// NostARHeadAttention_31344671326530
// MI455X (gfx1250) — matrix operand order repaired and checked by reading, not compiled, not run on hardware
//
#include <hip/hip_runtime.h>
#include <hip/hip_bf16.h>

// Problem constants (from reference)
#define B 16
#define S 2048
#define D 2048
#define H 16
#define HD 128
#define PAD_ID 50257
#define LN_EPS 1e-5f

typedef __bf16 bf16;
typedef __attribute__((ext_vector_type(16))) __bf16 v16bf;
typedef __attribute__((ext_vector_type(8)))  __bf16 v8bf;   // 16-byte vector load unit
typedef __attribute__((ext_vector_type(8)))  float  v8f;

// ---------------------------------------------------------------------------
// 1) LayerNorm: x[B*S, D] fp32 -> lnx bf16 (single pass over 256 MB input)
// ---------------------------------------------------------------------------
__global__ void ln_kernel(const float* __restrict__ x, const float* __restrict__ g,
                          const float* __restrict__ beta, bf16* __restrict__ lnx) {
    __shared__ float red[256];
    const int row = blockIdx.x;                // 0..B*S-1
    const size_t base = (size_t)row * D;
    const int t = threadIdx.x;                 // 256 threads, 8 elems each
    float v[8];
    float s = 0.f;
#pragma unroll
    for (int i = 0; i < 8; i++) { v[i] = x[base + i * 256 + t]; s += v[i]; }
    red[t] = s; __syncthreads();
    for (int off = 128; off > 0; off >>= 1) { if (t < off) red[t] += red[t + off]; __syncthreads(); }
    const float mu = red[0] * (1.f / D);
    __syncthreads();
    float sq = 0.f;
#pragma unroll
    for (int i = 0; i < 8; i++) { float c = v[i] - mu; sq += c * c; }
    red[t] = sq; __syncthreads();
    for (int off = 128; off > 0; off >>= 1) { if (t < off) red[t] += red[t + off]; __syncthreads(); }
    const float rstd = rsqrtf(red[0] * (1.f / D) + LN_EPS);
#pragma unroll
    for (int i = 0; i < 8; i++) {
        const int d = i * 256 + t;
        lnx[base + d] = (bf16)((v[i] - mu) * rstd * g[d] + beta[d]);
    }
}

// ---------------------------------------------------------------------------
// 2) Last non-pad token index per batch (argmax of cumsum == last non-pad)
// ---------------------------------------------------------------------------
__global__ void last_idx_kernel(const int* __restrict__ ids, int* __restrict__ idx) {
    __shared__ int red[256];
    const int b = blockIdx.x, t = threadIdx.x;
    int m = -1;
    for (int s = t; s < S; s += 256)
        if (ids[(size_t)b * S + s] != PAD_ID) m = s;   // s increasing -> keeps max
    red[t] = m; __syncthreads();
    for (int off = 128; off > 0; off >>= 1) { if (t < off) red[t] = max(red[t], red[t + off]); __syncthreads(); }
    if (t == 0) idx[b] = max(red[0], 0);               // all-pad -> 0 (matches argmax of zeros)
}

// ---------------------------------------------------------------------------
// 3) q[b, h*HD+hd] = lnx[b, idx[b], :] @ wq + bq   (tiny: 16x2048 output)
// ---------------------------------------------------------------------------
__global__ void proj_q_kernel(const bf16* __restrict__ lnx, const int* __restrict__ idx,
                              const float* __restrict__ wq, const float* __restrict__ bq,
                              float* __restrict__ q) {
    const int gid = blockIdx.x * 256 + threadIdx.x;    // B*D = 32768
    const int b = gid >> 11, j = gid & 2047;
    const bf16* xr = lnx + (size_t)(b * S + idx[b]) * D;
    float acc = bq[j];
    for (int d = 0; d < D; ++d) acc += (float)xr[d] * wq[(size_t)d * D + j];  // coalesced wq
    q[gid] = acc;
}

// ---------------------------------------------------------------------------
// 4) qk[b,h,d] = sum_hd q[b,h,hd] * wk[d, h*HD+hd]   -> bf16 [b][h][d]
//    (bk dropped: constant per (b,h) logit shift, softmax-invariant)
// ---------------------------------------------------------------------------
__global__ void make_qk_kernel(const float* __restrict__ q, const float* __restrict__ wk,
                               bf16* __restrict__ qkb) {
    const int gid = blockIdx.x * 256 + threadIdx.x;    // B*H*D = 524288
    const int b = gid >> 15;
    const int hd_ = gid & 32767;
    const int h = hd_ >> 11;
    const int d = hd_ & 2047;
    const float* qr = q + (size_t)b * D + h * HD;
    const float* wr = wk + (size_t)d * D + h * HD;
    float acc = 0.f;
#pragma unroll 4
    for (int k = 0; k < HD; ++k) acc += qr[k] * wr[k];
    qkb[gid] = (bf16)acc;
}

// ---------------------------------------------------------------------------
// 5) WMMA pass A: W[b,h,s] = sum_d lnx[b,s,d] * qk[b,h,d]
//    Per wave: one 16(s) x 16(h) tile, K-loop over D in steps of 32.
//    A/B layouts follow CDNA5 ISA 7.12.2 (16-bit A 16x32, B lane-column).
// ---------------------------------------------------------------------------
__global__ void logits_wmma_kernel(const bf16* __restrict__ lnx, const bf16* __restrict__ qkb,
                                   float* __restrict__ W) {
    const int b = blockIdx.y;
    const int s0 = blockIdx.x * 16;
    const int lane = threadIdx.x;                      // wave32
    const int r = lane & 15, seg = lane >> 4;
    const bf16* arow = lnx + ((size_t)b * S + s0 + r) * D + seg * 8;   // A: row m=s0+r
    const bf16* brow = qkb + ((size_t)b * H + r) * D + seg * 8;        // B: col n=h=r
    v8f acc = {};
    for (int k0 = 0; k0 < D; k0 += 32) {
        // A lane holds K = k0 + seg*8 + {0..7}  and  k0 + seg*8 + {16..23}
        v8bf a_lo = *(const v8bf*)(arow + k0);
        v8bf a_hi = *(const v8bf*)(arow + k0 + 16);
        // B lane holds K = k0 + seg*8 + {0..7}  and  k0 + seg*8 + {16..23} of column h
        v8bf b_lo = *(const v8bf*)(brow + k0);
        v8bf b_hi = *(const v8bf*)(brow + k0 + 16);
        v16bf a, bb;
#pragma unroll
        for (int i = 0; i < 8; i++) { a[i] = a_lo[i]; a[i + 8] = a_hi[i]; bb[i] = b_lo[i]; bb[i + 8] = b_hi[i]; }
        acc = __builtin_amdgcn_wmma_f32_16x16x32_bf16(false, a, false, bb, (short)0, acc,
                                                      false, false);
    }
    // D layout: acc[j] = D[M = j + 8*seg (=s offset), N = r (=h)] -> 8 contiguous s
    float* out = W + ((size_t)b * H + r) * S + s0 + seg * 8;
#pragma unroll
    for (int j = 0; j < 8; j++) out[j] = acc[j];
}

// ---------------------------------------------------------------------------
// 6) Softmax over s for each (b,h) row; emit normalized weights in bf16
// ---------------------------------------------------------------------------
__global__ void softmax_kernel(const float* __restrict__ W, bf16* __restrict__ Wn) {
    __shared__ float red[256];
    const int row = blockIdx.x;                        // B*H = 256 rows
    const float* wr = W + (size_t)row * S;
    const int t = threadIdx.x;
    float lv[8];
    float m = -1e30f;
#pragma unroll
    for (int i = 0; i < 8; i++) { lv[i] = wr[i * 256 + t]; m = fmaxf(m, lv[i]); }
    red[t] = m; __syncthreads();
    for (int off = 128; off > 0; off >>= 1) { if (t < off) red[t] = fmaxf(red[t], red[t + off]); __syncthreads(); }
    const float mx = red[0]; __syncthreads();
    float s = 0.f;
#pragma unroll
    for (int i = 0; i < 8; i++) { lv[i] = __expf(lv[i] - mx); s += lv[i]; }
    red[t] = s; __syncthreads();
    for (int off = 128; off > 0; off >>= 1) { if (t < off) red[t] += red[t + off]; __syncthreads(); }
    const float inv = 1.f / red[0];
    bf16* outr = Wn + (size_t)row * S;
#pragma unroll
    for (int i = 0; i < 8; i++) outr[i * 256 + t] = (bf16)(lv[i] * inv);
}

// ---------------------------------------------------------------------------
// 7) WMMA pass B: y[b,h,d] = sum_s Wn[b,h,s] * lnx[b,s,d]
//    Per wave: 16(h) x 16(d) tile, K-loop over S. B operand (lnx columns) is
//    gathered with strided b16 loads; L2-resident lines keep HBM traffic ~1x.
// ---------------------------------------------------------------------------
__global__ void y_wmma_kernel(const bf16* __restrict__ Wn, const bf16* __restrict__ lnx,
                              float* __restrict__ y) {
    const int b = blockIdx.y;
    const int d0 = blockIdx.x * 16;
    const int lane = threadIdx.x;
    const int r = lane & 15, seg = lane >> 4;
    const bf16* Ab = Wn + ((size_t)b * H + r) * S + seg * 8;    // A row m=h=r over K=s
    const bf16* Xb = lnx + (size_t)b * S * D + d0 + r;          // B column n=d0+r
    v8f acc = {};
    for (int k0 = 0; k0 < S; k0 += 32) {
        v8bf a_lo = *(const v8bf*)(Ab + k0);
        v8bf a_hi = *(const v8bf*)(Ab + k0 + 16);
        v16bf a, bb;
        const bf16* xp = Xb + (size_t)(k0 + seg * 8) * D;
#pragma unroll
        for (int kk = 0; kk < 8; ++kk) { bb[kk] = xp[(size_t)kk * D]; bb[kk + 8] = xp[(size_t)(kk + 16) * D]; }
#pragma unroll
        for (int i = 0; i < 8; i++) { a[i] = a_lo[i]; a[i + 8] = a_hi[i]; }
        acc = __builtin_amdgcn_wmma_f32_16x16x32_bf16(false, a, false, bb, (short)0, acc,
                                                      false, false);
    }
    // acc[j] = D[M = h = j + 8*seg, N = d0 + r]
#pragma unroll
    for (int j = 0; j < 8; j++)
        y[((size_t)b * H + j + 8 * seg) * D + d0 + r] = acc[j];
}

// ---------------------------------------------------------------------------
// 8) attn[b, h*HD+hd] = sum_d y[b,h,d] * wv[d, h*HD+hd] + bv  (softmax sums to 1)
// ---------------------------------------------------------------------------
__global__ void attn_kernel(const float* __restrict__ y, const float* __restrict__ wv,
                            const float* __restrict__ bv, float* __restrict__ attn) {
    const int gid = blockIdx.x * 256 + threadIdx.x;    // B*D
    const int b = gid >> 11, j = gid & 2047, h = j >> 7;
    const float* yr = y + ((size_t)b * H + h) * D;
    float acc = bv[j];
    for (int d = 0; d < D; ++d) acc += yr[d] * wv[(size_t)d * D + j];  // coalesced wv
    attn[gid] = acc;
}

// ---------------------------------------------------------------------------
// 9) out[b,p] = attn[b,:] @ wo + bo
// ---------------------------------------------------------------------------
__global__ void out_kernel(const float* __restrict__ attn, const float* __restrict__ wo,
                           const float* __restrict__ bo, float* __restrict__ out) {
    const int gid = blockIdx.x * 256 + threadIdx.x;    // B*PROJ
    const int b = gid >> 11, p = gid & 2047;
    const float* ar = attn + (size_t)b * D;
    float acc = bo[p];
    for (int d = 0; d < D; ++d) acc += ar[d] * wo[(size_t)d * D + p];  // coalesced wo
    out[gid] = acc;
}

// ---------------------------------------------------------------------------
extern "C" void kernel_launch(void* const* d_in, const int* in_sizes, int n_in,
                              void* d_out, int out_size, void* d_ws, size_t ws_size,
                              hipStream_t stream) {
    const float* x   = (const float*)d_in[0];
    const int*   ids = (const int*)  d_in[1];
    const float* lnw = (const float*)d_in[2];
    const float* lnb = (const float*)d_in[3];
    const float* wq  = (const float*)d_in[4];
    const float* bq  = (const float*)d_in[5];
    const float* wk  = (const float*)d_in[6];
    // d_in[7] = bk: adds a per-(b,h) constant to all logits -> softmax-invariant.
    const float* wv  = (const float*)d_in[8];
    const float* bv  = (const float*)d_in[9];
    const float* wo  = (const float*)d_in[10];
    const float* bo  = (const float*)d_in[11];
    float* out = (float*)d_out;

    // Workspace carving (256B aligned); total ~141 MB.
    char* ws = (char*)d_ws;
    size_t off = 0;
    auto carve = [&](size_t bytes) -> void* {
        void* p = ws + off;
        off += (bytes + 255) & ~(size_t)255;
        return p;
    };
    bf16*  lnx   = (bf16*) carve((size_t)B * S * D * sizeof(bf16));   // 128 MiB
    bf16*  qkb   = (bf16*) carve((size_t)B * H * D * sizeof(bf16));   // 1 MiB
    float* W     = (float*)carve((size_t)B * H * S * sizeof(float));  // 2 MiB
    bf16*  Wn    = (bf16*) carve((size_t)B * H * S * sizeof(bf16));   // 1 MiB
    float* qf    = (float*)carve((size_t)B * D * sizeof(float));      // 128 KiB
    float* yf    = (float*)carve((size_t)B * H * D * sizeof(float));  // 2 MiB
    float* attnf = (float*)carve((size_t)B * D * sizeof(float));      // 128 KiB
    int*   idx   = (int*)  carve(B * sizeof(int));
    (void)ws_size; (void)in_sizes; (void)n_in; (void)out_size;

    ln_kernel        <<<B * S, 256, 0, stream>>>(x, lnw, lnb, lnx);
    last_idx_kernel  <<<B, 256, 0, stream>>>(ids, idx);
    proj_q_kernel    <<<(B * D) / 256, 256, 0, stream>>>(lnx, idx, wq, bq, qf);
    make_qk_kernel   <<<(B * H * D) / 256, 256, 0, stream>>>(qf, wk, qkb);
    logits_wmma_kernel<<<dim3(S / 16, B), 32, 0, stream>>>(lnx, qkb, W);
    softmax_kernel   <<<B * H, 256, 0, stream>>>(W, Wn);
    y_wmma_kernel    <<<dim3(D / 16, B), 32, 0, stream>>>(Wn, lnx, yf);
    attn_kernel      <<<(B * D) / 256, 256, 0, stream>>>(yf, wv, bv, attnf);
    out_kernel       <<<(B * D) / 256, 256, 0, stream>>>(attnf, wo, bo, out);
}
